// DifferenceOfGaussians_32641751450131
// MI455X (gfx1250) — matrix operand order repaired and checked by reading, not compiled, not run on hardware
//
#include <hip/hip_runtime.h>
#include <hip/hip_bf16.h>

typedef __attribute__((ext_vector_type(16))) _Float16 v16h;
typedef __attribute__((ext_vector_type(8)))  float    v8f;
typedef unsigned int u32x4 __attribute__((ext_vector_type(4)));
typedef int          i32x8 __attribute__((ext_vector_type(8)));
typedef int          i32x4 __attribute__((ext_vector_type(4)));

#define IMG_H 1024
#define IMG_W 1024
#define NSCALE 14      // k+1 Gaussian scales
#define NDOG   13      // DoG levels
#define RAD    53      // max radius
#define KSZ    107     // 2*RAD+1
#define KPAD   128     // K dim for 4 x K=32 WMMA (covers 107 taps + 15 tile offset)
#define UOFF   16      // leading zero pad so Toeplitz index (k - m) never underflows
#define USTR   160     // padded per-scale 1D kernel stride: 16 zeros + 128 (+16 tail)

// ---------------------------------------------------------------------------
// Stage 0: recover normalized 1D separable kernels from the 2D weights.
// W2d[s] = outer(u,u) with sum==1, so u[j] = W2d[s][53][j] / sum_j W2d[s][53][j].
// Stored f16, with 16 leading zeros (Toeplitz underflow pad) and zero tail.
// ---------------------------------------------------------------------------
__global__ __launch_bounds__(32)
void dog_prep_kernels(const float* __restrict__ w2d, _Float16* __restrict__ u16) {
    int s = threadIdx.x;
    if (s < NSCALE) {
        const float* row = w2d + (size_t)s * KSZ * KSZ + (size_t)RAD * KSZ;
        float sum = 0.0f;
        for (int j = 0; j < KSZ; ++j) sum += row[j];
        float inv = 1.0f / sum;
        for (int j = 0; j < USTR; ++j) {
            int t = j - UOFF;
            float v = (t >= 0 && t < KSZ) ? row[t] * inv : 0.0f;
            u16[s * USTR + j] = (_Float16)v;
        }
    }
}

// ---------------------------------------------------------------------------
// Stage 1: horizontal blur, all 14 scales, one 16x16 tile per wave.
// C(16x16) = A(16x128 image tile, f16) x B(128x16 Toeplitz kernel, f16)
// via 4 x v_wmma_f32_16x16x32_f16. A fragments are shared across scales.
// ---------------------------------------------------------------------------
__global__ __launch_bounds__(32)
void dog_hblur(const float* __restrict__ img,
               const _Float16* __restrict__ uAll,
               _Float16* __restrict__ hOut) {
    __shared__ _Float16 uLds[NSCALE * USTR];     // 4.4 KB
    __shared__ _Float16 tile[16][KPAD];          // 4.0 KB: rows y0..y0+15, cols x0-53..x0+74

    const int l  = threadIdx.x;
    const int x0 = blockIdx.x * 16;
    const int y0 = blockIdx.y * 16;
    const int M  = l & 15;
    const int hi = l >> 4;                       // 0 for lanes 0-15, 1 for 16-31

    for (int t = l; t < NSCALE * USTR; t += 32) uLds[t] = uAll[t];
    for (int t = l; t < 16 * KPAD; t += 32) {
        int r = t >> 7, c = t & (KPAD - 1);
        int gx = x0 - RAD + c;
        float v = (gx >= 0 && gx < IMG_W) ? img[(size_t)(y0 + r) * IMG_W + gx] : 0.0f;
        tile[r][c] = (_Float16)v;
    }
    __syncthreads();

    // Build the 4 A fragments once (image is identical for every scale).
    v16h A[4];
#pragma unroll
    for (int c = 0; c < 4; ++c) {
#pragma unroll
        for (int i = 0; i < 16; ++i) {
            int K = hi * 16 + i;                             // contiguous k = 16*hi + i (same order as B)
            A[c][i] = tile[M][c * 32 + K];
        }
    }

    for (int s = 0; s < NSCALE; ++s) {
        const _Float16* us = &uLds[s * USTR + UOFF - M];     // Toeplitz base for this lane
        v8f C = {};
#pragma unroll
        for (int c = 0; c < 4; ++c) {
            v16h B;
#pragma unroll
            for (int i = 0; i < 16; ++i) {
                int k = c * 32 + hi * 16 + i;                // contiguous k order, same as A
                B[i] = us[k];                                // B[k,n] = u[k-n], pad handles k<n
            }
            C = __builtin_amdgcn_wmma_f32_16x16x32_f16(
                    false, A[c], false, B, (short)0, C, false, false);
        }
        _Float16* hs = hOut + (size_t)s * IMG_H * IMG_W;
#pragma unroll
        for (int e = 0; e < 8; ++e) {                        // C layout: M = e + 8*hi, N = lane%16
            int row = y0 + hi * 8 + e;
            hs[(size_t)row * IMG_W + x0 + M] = (_Float16)C[e];
        }
    }
}

// ---------------------------------------------------------------------------
// Stage 2: vertical blur fused with DoG. One 16x16 tile per wave, scales
// processed sequentially so g[s-1] stays in VGPRs for the subtraction.
// C(16x16) = A(16x128 Toeplitz u_s) x B(128x16 column tile of h[s]).
// Interior tiles stage the 128x16 f16 h-tile via the Tensor Data Mover
// (tensor_load_to_lds + s_wait_tensorcnt); edge tiles stage manually.
// dog[s-1] = (g[s-1] - g[s]) * sigma[s-1] written straight to d_out.
// ---------------------------------------------------------------------------
__global__ __launch_bounds__(32)
void dog_vblur_dog(const _Float16* __restrict__ hIn,
                   const _Float16* __restrict__ uAll,
                   const float* __restrict__ sigmas,
                   float* __restrict__ dogOut) {
    __shared__ _Float16 uLds[NSCALE * USTR];     // 4.4 KB
    __shared__ _Float16 colT[KPAD][16];          // 4.0 KB: rows y0-53..y0+74, cols x0..x0+15

    const int l  = threadIdx.x;
    const int x0 = blockIdx.x * 16;
    const int y0 = blockIdx.y * 16;
    const int M  = l & 15;
    const int hi = l >> 4;

    for (int t = l; t < NSCALE * USTR; t += 32) uLds[t] = uAll[t];

    const bool interior = (y0 - RAD >= 0) && (y0 - RAD + KPAD <= IMG_H);  // uniform
    const unsigned ldsAddr = (unsigned)(uintptr_t)&colT[0][0];  // low 32b = LDS byte offset

    v8f Cprev = {};
    for (int s = 0; s < NSCALE; ++s) {
        const _Float16* hs = hIn + (size_t)s * IMG_H * IMG_W;
        if (s + 1 < NSCALE)   // warm L2 for the next scale's tile (global_prefetch_b8)
            __builtin_prefetch(hs + (size_t)IMG_H * IMG_W + (size_t)y0 * IMG_W + x0, 0, 1);
        __syncthreads();
        if (interior) {
            // TDM: 2D D# -> load tile_dim0=16 elems x tile_dim1=128 rows of f16
            unsigned long long gaddr =
                (unsigned long long)(uintptr_t)(hs + (size_t)(y0 - RAD) * IMG_W + x0);
            u32x4 g0 = { 0x1u,                                   // count=1, user desc
                         ldsAddr,                                // D#.lds_addr
                         (unsigned)gaddr,                        // D#.global_addr[31:0]
                         ((unsigned)(gaddr >> 32) & 0x01FFFFFFu)
                           | 0x80000000u };                      // addr[56:32] | type=2
            i32x8 g1 = { 0x00010000,                             // data_size=1 (2 bytes)
                         (int)(1024u << 16),                     // tensor_dim0 = 1024
                         (int)(1024u << 16),                     // tensor_dim1 = 1024
                         (int)(16u  << 16),                      // tile_dim0 = 16
                         128,                                    // tile_dim1 = 128
                         1024,                                   // tensor_dim0_stride lo32
                         0,                                      // stride0 hi | stride1 lo16
                         0x10 };                                 // tensor_dim1_stride = 1M
            i32x4 z4 = { 0, 0, 0, 0 };
            i32x8 z8 = { 0, 0, 0, 0, 0, 0, 0, 0 };
            __builtin_amdgcn_tensor_load_to_lds(g0, g1, z4, z4, z8, 0);
            __builtin_amdgcn_s_wait_tensorcnt(0);
        } else {
            for (int t = l; t < KPAD * 16; t += 32) {
                int r = t >> 4, cc = t & 15;
                int gy = y0 - RAD + r;
                colT[r][cc] = (gy >= 0 && gy < IMG_H)
                              ? hs[(size_t)gy * IMG_W + x0 + cc] : (_Float16)0.0f;
            }
        }
        __syncthreads();

        const _Float16* us = &uLds[s * USTR + UOFF - M];     // Toeplitz base for this lane
        v8f C = {};
#pragma unroll
        for (int c = 0; c < 4; ++c) {
            v16h A, B;
#pragma unroll
            for (int i = 0; i < 16; ++i) {
                int K = hi * 16 + i;
                A[i] = us[c * 32 + K];                       // A[m,k] = u_s[k-m]
                int k = c * 32 + hi * 16 + i;
                B[i] = colT[k][M];
            }
            C = __builtin_amdgcn_wmma_f32_16x16x32_f16(
                    false, A, false, B, (short)0, C, false, false);
        }

        if (s > 0) {
            float sg = sigmas[s - 1];
            float* dd = dogOut + (size_t)(s - 1) * IMG_H * IMG_W;
#pragma unroll
            for (int e = 0; e < 8; ++e) {
                int row = y0 + hi * 8 + e;
                dd[(size_t)row * IMG_W + x0 + M] = (Cprev[e] - C[e]) * sg;
            }
        }
        Cprev = C;
    }
}

// ---------------------------------------------------------------------------
// Stage 3: 3x3x3 max over (scale,y,x) with -inf padding; mask = (dog==max)&&(dog>1e-3)
// ---------------------------------------------------------------------------
__global__ void dog_maxpool_mask(const float* __restrict__ dog, float* __restrict__ mask) {
    size_t idx = (size_t)blockIdx.x * blockDim.x + threadIdx.x;
    const size_t total = (size_t)NDOG * IMG_H * IMG_W;
    if (idx >= total) return;
    int x = (int)(idx & (IMG_W - 1));
    int y = (int)((idx >> 10) & (IMG_H - 1));
    int s = (int)(idx >> 20);
    float v = dog[idx];
    float m = -INFINITY;
    for (int dz = -1; dz <= 1; ++dz) {
        int ss = s + dz;
        if (ss < 0 || ss >= NDOG) continue;
        for (int dy = -1; dy <= 1; ++dy) {
            int yy = y + dy;
            if (yy < 0 || yy >= IMG_H) continue;
            for (int dx = -1; dx <= 1; ++dx) {
                int xx = x + dx;
                if (xx < 0 || xx >= IMG_W) continue;
                float w = dog[((size_t)ss << 20) | ((size_t)yy << 10) | (size_t)xx];
                m = fmaxf(m, w);
            }
        }
    }
    mask[idx] = (v == m && v > 0.001f) ? 1.0f : 0.0f;
}

// ---------------------------------------------------------------------------
extern "C" void kernel_launch(void* const* d_in, const int* in_sizes, int n_in,
                              void* d_out, int out_size, void* d_ws, size_t ws_size,
                              hipStream_t stream) {
    const float* img    = (const float*)d_in[0];   // [1,1,1024,1024]
    const float* w2d    = (const float*)d_in[1];   // [14,1,107,107]
    const float* sigmas = (const float*)d_in[2];   // [14]

    float* dogOut  = (float*)d_out;                          // 13*1024*1024 f32
    float* maskOut = (float*)d_out + (size_t)NDOG * IMG_H * IMG_W;

    // workspace: [0, 14*160*2) padded 1D kernels f16 ; [8192, +28MB) h intermediate f16
    _Float16* u16 = (_Float16*)d_ws;
    _Float16* h16 = (_Float16*)((char*)d_ws + 8192);

    dog_prep_kernels<<<1, 32, 0, stream>>>(w2d, u16);

    dim3 tiles(IMG_W / 16, IMG_H / 16);
    dog_hblur<<<tiles, 32, 0, stream>>>(img, u16, h16);
    dog_vblur_dog<<<tiles, 32, 0, stream>>>(h16, u16, sigmas, dogOut);

    size_t total = (size_t)NDOG * IMG_H * IMG_W;
    dog_maxpool_mask<<<(unsigned)((total + 255) / 256), 256, 0, stream>>>(dogOut, maskOut);
}
